// MultiHeadAttention_45844480918317
// MI455X (gfx1250) — matrix operand order repaired and checked by reading, not compiled, not run on hardware
//
#include <hip/hip_runtime.h>
#include <hip/hip_bf16.h>

// ---------------------------------------------------------------------------
// MHA forward for B=2, T=2048, D=1024, H=16, DH=64 on gfx1250 (wave32, WMMA).
// Pipeline: 3x bf16-WMMA projection GEMMs (async-LDS staged, double buffered)
//           -> RoPE -> flash attention (WMMA, V stored transposed)
//           -> bf16-WMMA output projection (f32 out).
// ---------------------------------------------------------------------------

typedef __attribute__((ext_vector_type(16))) __bf16 v16bf;
typedef __attribute__((ext_vector_type(8)))  float  v8f;
typedef int v4i __attribute__((vector_size(16)));

#define WMMA_BF16(a, b, c) \
  __builtin_amdgcn_wmma_f32_16x16x32_bf16(false, (a), false, (b), (short)0, (c), false, false)

#define MHA_B   2
#define MHA_T   2048
#define MHA_D   1024
#define MHA_H   16
#define MHA_DH  64

#if defined(__has_builtin)
#if __has_builtin(__builtin_amdgcn_global_load_async_to_lds_b128)
#define HAS_ASYNC_LDS 1
#endif
#endif
#ifndef HAS_ASYNC_LDS
#define HAS_ASYNC_LDS 0
#endif

#define AS1 __attribute__((address_space(1)))
#define AS3 __attribute__((address_space(3)))

static __device__ inline void wait_async_all() {
#if HAS_ASYNC_LDS
#if defined(__has_builtin) && __has_builtin(__builtin_amdgcn_s_wait_asynccnt)
  __builtin_amdgcn_s_wait_asynccnt(0);
#else
  asm volatile("s_wait_asynccnt 0x0" ::: "memory");
#endif
#endif
}

// 16-byte copy global -> LDS (async if available, else synchronous via regs)
static __device__ inline void cp16_g2l(const void* g, void* l) {
#if HAS_ASYNC_LDS
  __builtin_amdgcn_global_load_async_to_lds_b128((AS1 v4i*)g, (AS3 v4i*)l, 0, 0);
#else
  *(int4*)l = *(const int4*)g;
#endif
}

// ---- global fragment loaders (used by attention) ---------------------------
// A fragment (16x32, bf16): lane row = lane&15; elems 0..7 -> K=kb..kb+7,
// elems 8..15 -> K=16+kb..16+kb+7 with kb = (lane>>4)*8.  (ISA 7.12.2)
static __device__ inline v16bf frag_a_g(const __bf16* A, int lda, int row0, int k0, int lane) {
  int row = row0 + (lane & 15);
  int kb  = k0 + ((lane >> 4) << 3);
  const __bf16* p = A + (size_t)row * lda + kb;
  v16bf f;
#pragma unroll
  for (int i = 0; i < 8; ++i) { f[i] = p[i]; f[i + 8] = p[i + 16]; }
  return f;
}

// B fragment (32x16, bf16): lane column = lane&15; elems 0..7 -> K=kb..kb+7,
// elems 8..15 -> K=16+kb..16+kb+7 with kb = (lane>>4)*8.
// B = M^T: column n of B is row n of M (row-major, contiguous K).
// Used for K (S = Q K^T) and for V^T (O += P V with V stored transposed).
static __device__ inline v16bf frag_bt_g(const __bf16* M, int ldm, int col0, int k0, int lane) {
  int col = col0 + (lane & 15);
  int kb  = k0 + ((lane >> 4) << 3);
  const __bf16* p = M + (size_t)col * ldm + kb;
  v16bf f;
#pragma unroll
  for (int i = 0; i < 8; ++i) { f[i] = p[i]; f[i + 8] = p[i + 16]; }
  return f;
}

// ---- LDS fragment loaders (GEMM); tiles stored row-major, 32 elems/row -----
template <typename TA>
static __device__ inline v16bf frag_a_lds(const TA* base, int row0, int lane) {
  int row = row0 + (lane & 15);
  int kb  = (lane >> 4) << 3;
  const TA* p = base + row * 32 + kb;
  v16bf f;
#pragma unroll
  for (int i = 0; i < 8; ++i) { f[i] = (__bf16)p[i]; f[i + 8] = (__bf16)p[i + 16]; }
  return f;
}
static __device__ inline v16bf frag_b_lds(const float* base, int col0, int lane) {
  int col = col0 + (lane & 15);
  int kb  = (lane >> 4) << 3;
  const float* p = base + col * 32 + kb;
  v16bf f;
#pragma unroll
  for (int i = 0; i < 8; ++i) { f[i] = (__bf16)p[i]; f[i + 8] = (__bf16)p[i + 16]; }
  return f;
}

// ---- GEMM: Y = X @ W^T  (X: MxK TA, W: NxK f32, row-major) -----------------
// Block: 256 thr = 8 waves (2x4); block tile 64x128; wave tile 32x32 (4 WMMA).
// K-step 32; X/W tiles double-buffered in LDS via async global->LDS copies.
// layout: 0 = row-major [M,N]; 1 = heads [B,H,T,DH]; 2 = heads transposed
//         [B,H,DH,T] (used for V so attention P@V reads contiguous K).
template <typename TA>
__global__ __launch_bounds__(256) void gemm_xwt_kernel(
    const TA* __restrict__ X, const float* __restrict__ W,
    __bf16* __restrict__ Ybf, float* __restrict__ Yf,
    int M, int N, int K, int layout) {
  __shared__ __align__(16) TA    lx[2][64 * 32];
  __shared__ __align__(16) float lw[2][128 * 32];

  const int tid   = threadIdx.x;
  const int lane  = tid & 31;
  const int wave  = tid >> 5;
  const int wrow  = wave >> 2;              // 0..1
  const int wcol  = wave & 3;               // 0..3
  const int mtile = blockIdx.y * 64;
  const int ntile = blockIdx.x * 128;

  constexpr int XROWB  = 32 * (int)sizeof(TA);
  constexpr int XTILEB = 64 * XROWB;
  constexpr int WTILEB = 128 * 128;

  auto stage = [&](int k0, int buf) {
#pragma unroll 2
    for (int ofs = tid * 16; ofs < XTILEB; ofs += 256 * 16) {
      int row  = ofs / XROWB;
      int colb = ofs % XROWB;
      const char* g = (const char*)(X + (size_t)(mtile + row) * K + k0) + colb;
      cp16_g2l(g, (char*)&lx[buf][0] + ofs);
    }
#pragma unroll 4
    for (int ofs = tid * 16; ofs < WTILEB; ofs += 256 * 16) {
      int row  = ofs >> 7;
      int colb = ofs & 127;
      const char* g = (const char*)(W + (size_t)(ntile + row) * K + k0) + colb;
      cp16_g2l(g, (char*)&lw[buf][0] + ofs);
    }
  };

  v8f acc[2][2];
#pragma unroll
  for (int i = 0; i < 2; ++i)
#pragma unroll
    for (int j = 0; j < 2; ++j) acc[i][j] = (v8f){};

  const int nk = K / 32;
  stage(0, 0);
  for (int kk = 0; kk < nk; ++kk) {
    wait_async_all();
    __syncthreads();
    if (kk + 1 < nk) stage((kk + 1) * 32, (kk + 1) & 1);

    const TA*    xb = &lx[kk & 1][0];
    const float* wb = &lw[kk & 1][0];
    v16bf a0 = frag_a_lds(xb, wrow * 32 + 0,  lane);
    v16bf a1 = frag_a_lds(xb, wrow * 32 + 16, lane);
    v16bf b0 = frag_b_lds(wb, wcol * 32 + 0,  lane);
    v16bf b1 = frag_b_lds(wb, wcol * 32 + 16, lane);
    acc[0][0] = WMMA_BF16(a0, b0, acc[0][0]);
    acc[0][1] = WMMA_BF16(a0, b1, acc[0][1]);
    acc[1][0] = WMMA_BF16(a1, b0, acc[1][0]);
    acc[1][1] = WMMA_BF16(a1, b1, acc[1][1]);
  }

  // store 4 16x16 tiles (C/D layout: row = r + 8*(lane>>4), col = lane&15)
#pragma unroll
  for (int i = 0; i < 2; ++i)
#pragma unroll
    for (int j = 0; j < 2; ++j) {
      int n     = ntile + wcol * 32 + j * 16 + (lane & 15);
      int mbase = mtile + wrow * 32 + i * 16 + ((lane >> 4) << 3);
#pragma unroll
      for (int r = 0; r < 8; ++r) {
        int m = mbase + r;
        size_t off;
        if (layout == 1) {
          int bb = m >> 11, t = m & (MHA_T - 1);
          int h  = n >> 6,  d = n & (MHA_DH - 1);
          off = (((size_t)bb * MHA_H + h) * MHA_T + t) * MHA_DH + d;
        } else if (layout == 2) {
          int bb = m >> 11, t = m & (MHA_T - 1);
          int h  = n >> 6,  d = n & (MHA_DH - 1);
          off = (((size_t)bb * MHA_H + h) * MHA_DH + d) * MHA_T + t;
        } else {
          off = (size_t)m * N + n;
        }
        if (Ybf) Ybf[off] = (__bf16)acc[i][j][r];
        else     Yf[off]  = acc[i][j][r];
      }
    }
}

// ---- RoPE on Q and K in [B,H,T,DH] bf16 ------------------------------------
__global__ __launch_bounds__(256) void rope_kernel(__bf16* __restrict__ Q,
                                                   __bf16* __restrict__ K) {
  int i  = blockIdx.x * blockDim.x + threadIdx.x;   // B*H*T*32 threads
  int j  = i & 31;                                  // freq index 0..31
  int t  = (i >> 5) & (MHA_T - 1);
  int bh = i >> 16;
  size_t base = ((size_t)bh * MHA_T + t) * MHA_DH;
  float inv = __expf(-0.28782313662425572f * (float)j);  // 10000^(-j/32)
  float ang = (float)t * inv;
  float c = __cosf(ang), s = __sinf(ang);

  float q1 = (float)Q[base + j], q2 = (float)Q[base + j + 32];
  Q[base + j]      = (__bf16)(q1 * c - q2 * s);
  Q[base + j + 32] = (__bf16)(q2 * c + q1 * s);

  float k1 = (float)K[base + j], k2 = (float)K[base + j + 32];
  K[base + j]      = (__bf16)(k1 * c - k2 * s);
  K[base + j + 32] = (__bf16)(k2 * c + k1 * s);
}

// ---- Flash attention: per (b,h); wave owns a 16-row Q tile ------------------
// Q,K in [B,H,T,DH]; V in [B,H,DH,T] (transposed -> contiguous B fragments).
__global__ __launch_bounds__(128) void flash_attn_kernel(
    const __bf16* __restrict__ Q, const __bf16* __restrict__ K,
    const __bf16* __restrict__ VT, __bf16* __restrict__ O) {
  __shared__ __bf16 plds[4][16 * 32];

  int lane = threadIdx.x & 31;
  int wave = threadIdx.x >> 5;
  int bh   = blockIdx.y;                       // 0..B*H-1
  int qb   = blockIdx.x * 64 + wave * 16;      // query tile base

  const __bf16* Qh  = Q  + (size_t)bh * MHA_T * MHA_DH;
  const __bf16* Kh  = K  + (size_t)bh * MHA_T * MHA_DH;
  const __bf16* VTh = VT + (size_t)bh * MHA_T * MHA_DH;  // [DH, T]
  __bf16* pt = &plds[wave][0];

  v16bf aq0 = frag_a_g(Qh, MHA_DH, qb, 0, lane);
  v16bf aq1 = frag_a_g(Qh, MHA_DH, qb, 32, lane);

  v8f oacc[4];
#pragma unroll
  for (int c = 0; c < 4; ++c) oacc[c] = (v8f){};
  float mrow[8], lrow[8];
#pragma unroll
  for (int r = 0; r < 8; ++r) { mrow[r] = -3.0e38f; lrow[r] = 0.0f; }

  const int moff = (lane >> 4) << 3;
  const int ncol = lane & 15;
  const int nkb  = (qb + 47) >> 5;     // # of 32-wide key blocks (causal)

  for (int kb = 0; kb < nkb; ++kb) {
    int k0 = kb * 32;
    v8f s0 = {}, s1 = {};
    {
      v16bf b;
      b = frag_bt_g(Kh, MHA_DH, k0,      0,  lane); s0 = WMMA_BF16(aq0, b, s0);
      b = frag_bt_g(Kh, MHA_DH, k0,      32, lane); s0 = WMMA_BF16(aq1, b, s0);
      b = frag_bt_g(Kh, MHA_DH, k0 + 16, 0,  lane); s1 = WMMA_BF16(aq0, b, s1);
      b = frag_bt_g(Kh, MHA_DH, k0 + 16, 32, lane); s1 = WMMA_BF16(aq1, b, s1);
    }
#pragma unroll
    for (int r = 0; r < 8; ++r) {
      int row  = qb + moff + r;
      float x0 = s0[r] * 0.125f;               // 1/sqrt(64)
      float x1 = s1[r] * 0.125f;
      if (k0 + ncol      > row) x0 = -3.0e38f; // causal mask
      if (k0 + 16 + ncol > row) x1 = -3.0e38f;
      float mx = fmaxf(x0, x1);
#pragma unroll
      for (int off = 1; off < 16; off <<= 1) mx = fmaxf(mx, __shfl_xor(mx, off, 32));
      float mnew  = fmaxf(mrow[r], mx);
      float alpha = __expf(mrow[r] - mnew);
      float p0 = __expf(x0 - mnew);
      float p1 = __expf(x1 - mnew);
      float rs = p0 + p1;
#pragma unroll
      for (int off = 1; off < 16; off <<= 1) rs += __shfl_xor(rs, off, 32);
      lrow[r] = lrow[r] * alpha + rs;
      mrow[r] = mnew;
#pragma unroll
      for (int c = 0; c < 4; ++c) oacc[c][r] *= alpha;
      pt[(moff + r) * 32 + ncol]      = (__bf16)p0;
      pt[(moff + r) * 32 + 16 + ncol] = (__bf16)p1;
    }
    // restage P (D-layout) -> A fragment via per-wave LDS
    v16bf ap;
    {
      int row = lane & 15;
      int kbq = (lane >> 4) << 3;
#pragma unroll
      for (int i = 0; i < 8; ++i) {
        ap[i]     = pt[row * 32 + kbq + i];
        ap[i + 8] = pt[row * 32 + 16 + kbq + i];
      }
    }
    // O += P @ V : B fragment = rows of V^T (contiguous K), ldm = T
#pragma unroll
    for (int c = 0; c < 4; ++c) {
      v16bf bv = frag_bt_g(VTh, MHA_T, c * 16, k0, lane);
      oacc[c] = WMMA_BF16(ap, bv, oacc[c]);
    }
  }

  int bb = bh >> 4, h = bh & 15;
#pragma unroll
  for (int c = 0; c < 4; ++c) {
    int d = c * 16 + ncol;
#pragma unroll
    for (int r = 0; r < 8; ++r) {
      int m = qb + moff + r;
      float val = oacc[c][r] / lrow[r];
      O[(((size_t)bb * MHA_T + m) * MHA_H + h) * MHA_DH + d] = (__bf16)val;
    }
  }
}

// ---------------------------------------------------------------------------
extern "C" void kernel_launch(void* const* d_in, const int* in_sizes, int n_in,
                              void* d_out, int out_size, void* d_ws, size_t ws_size,
                              hipStream_t stream) {
  const float* x  = (const float*)d_in[0];
  const float* Wq = (const float*)d_in[1];
  const float* Wk = (const float*)d_in[2];
  const float* Wv = (const float*)d_in[3];
  const float* Wo = (const float*)d_in[4];
  // d_in[5] (mask) unused: causal mask applied analytically.
  float* out = (float*)d_out;

  const size_t elems = (size_t)MHA_B * MHA_H * MHA_T * MHA_DH;  // 4,194,304
  __bf16* qws = (__bf16*)d_ws;
  __bf16* kws = qws + elems;
  __bf16* vws = kws + elems;   // V stored transposed per head: [B,H,DH,T]
  __bf16* aws = vws + elems;   // attention output, [B,T,H,DH] == [B*T, D]

  const int M  = MHA_B * MHA_T;  // 4096
  const int N  = MHA_D;          // 1024
  const int Kd = MHA_D;          // 1024
  dim3 gblk(256);
  dim3 ggrid(N / 128, M / 64);   // (8, 64)

  gemm_xwt_kernel<float><<<ggrid, gblk, 0, stream>>>(x, Wq, qws, nullptr, M, N, Kd, 1);
  gemm_xwt_kernel<float><<<ggrid, gblk, 0, stream>>>(x, Wk, kws, nullptr, M, N, Kd, 1);
  gemm_xwt_kernel<float><<<ggrid, gblk, 0, stream>>>(x, Wv, vws, nullptr, M, N, Kd, 2);

  const int rope_threads = MHA_B * MHA_H * MHA_T * 32;  // 2,097,152
  rope_kernel<<<rope_threads / 256, 256, 0, stream>>>(qws, kws);

  flash_attn_kernel<<<dim3(MHA_T / 64, MHA_B * MHA_H), 128, 0, stream>>>(qws, kws, vws, aws);

  gemm_xwt_kernel<__bf16><<<ggrid, gblk, 0, stream>>>(aws, Wo, nullptr, out, M, N, Kd, 0);
}
